// Attention_data_51874615001679
// MI455X (gfx1250) — matrix operand order repaired and checked by reading, not compiled, not run on hardware
//
#include <hip/hip_runtime.h>

// MI455X (gfx1250) fused uncertainty-attention.
// Memory-bound (~1.2 GB traffic @ 23.3 TB/s >> 33 GFLOP of matmul), so all
// GEMMs run through v_wmma_f32_16x16x32_bf16 (f32 accum) and the [B,H,N,N]
// pointwise stage is fused into a single pass per (b,n) row block.
// K-loops are software-pipelined with a 2x-unrolled ping-pong fragment buffer:
// loads land directly in the registers the next WMMA consumes (no rotation
// copies), and the wave only takes partial s_wait_loadcnt before each WMMA.

#define BBATCH 4
#define NTOK   1024
#define CDIM   768
#define NHEAD  12
#define HDIM   64
#define NBH    (BBATCH * NHEAD)   // 48

typedef __attribute__((ext_vector_type(16))) __bf16         bf16x16;
typedef __attribute__((ext_vector_type(8)))  float          f32x8;
typedef __attribute__((ext_vector_type(16))) unsigned short u16x16;
typedef __attribute__((ext_vector_type(8)))  unsigned short u16x8;

__device__ __forceinline__ unsigned short f2bf(float f) {
  unsigned u = __builtin_bit_cast(unsigned, f);
  u += 0x7FFFu + ((u >> 16) & 1u);          // round-to-nearest-even
  return (unsigned short)(u >> 16);
}

// A fragment: 16x32 bf16, lane L serves row (L&15); lanes 0-15 hold K
// {0..15}, lanes 16-31 hold K {16..31} (contiguous, matching B).
// `rowk` points at A[row] + kbase (row-major, K contiguous).
__device__ __forceinline__ bf16x16 load_a_frag(const unsigned short* rowk, int lane) {
  u16x16 t = *(const u16x16*)(rowk + (lane >> 4) * 16);
  return __builtin_bit_cast(bf16x16, t);
}

// B fragment: 32x16 bf16, lane L serves column (L&15); lane half owns a
// contiguous 16-wide K chunk. `colk` points at W[col] + kbase (W row-major
// [N,K], i.e. computing A @ W^T).
__device__ __forceinline__ bf16x16 load_b_frag(const unsigned short* colk, int lane) {
  u16x16 t = *(const u16x16*)(colk + (lane >> 4) * 16);
  return __builtin_bit_cast(bf16x16, t);
}

__device__ __forceinline__ f32x8 wmma_bf16(bf16x16 a, bf16x16 b, f32x8 c) {
  return __builtin_amdgcn_wmma_f32_16x16x32_bf16(false, a, false, b, (short)0, c,
                                                 false, false);
}

// Software-pipelined 16xK x Kx64 tile MAC: A row-major, B as 4 column tiles
// of W[N,K] rows. 2x-unrolled ping-pong double buffering (K/32 must be even,
// true for all call sites: 768, 64, 1024).
__device__ __forceinline__ void gemm_tile_pipelined(const unsigned short* arow,
                                                    const unsigned short* const brow[4],
                                                    int K, int lane, f32x8 acc[4]) {
  bf16x16 a0 = load_a_frag(arow, lane);
  bf16x16 b0[4];
#pragma unroll
  for (int s = 0; s < 4; ++s) b0[s] = load_b_frag(brow[s], lane);
  int kk = 0;
  for (; kk < K - 64; kk += 64) {
    // load step k+1 into ping buffers, compute step k from pong buffers
    bf16x16 a1 = load_a_frag(arow + kk + 32, lane);
    bf16x16 b1[4];
#pragma unroll
    for (int s = 0; s < 4; ++s) b1[s] = load_b_frag(brow[s] + kk + 32, lane);
#pragma unroll
    for (int s = 0; s < 4; ++s) acc[s] = wmma_bf16(a0, b0[s], acc[s]);
    // load step k+2 back into pong buffers, compute step k+1
    a0 = load_a_frag(arow + kk + 64, lane);
#pragma unroll
    for (int s = 0; s < 4; ++s) b0[s] = load_b_frag(brow[s] + kk + 64, lane);
#pragma unroll
    for (int s = 0; s < 4; ++s) acc[s] = wmma_bf16(a1, b1[s], acc[s]);
  }
  // kk == K-64: step kk in pong buffers; load final step, drain both.
  bf16x16 a1 = load_a_frag(arow + kk + 32, lane);
  bf16x16 b1[4];
#pragma unroll
  for (int s = 0; s < 4; ++s) b1[s] = load_b_frag(brow[s] + kk + 32, lane);
#pragma unroll
  for (int s = 0; s < 4; ++s) acc[s] = wmma_bf16(a0, b0[s], acc[s]);
#pragma unroll
  for (int s = 0; s < 4; ++s) acc[s] = wmma_bf16(a1, b1[s], acc[s]);
}

// ---------------------------------------------------------------- convert --
__global__ __launch_bounds__(256) void k_convert(const float* __restrict__ x,
                                                 const float* __restrict__ qkvw,
                                                 const float* __restrict__ projw,
                                                 unsigned short* __restrict__ Xb,
                                                 unsigned short* __restrict__ Wqkvb,
                                                 unsigned short* __restrict__ Wpb) {
  const int nx = BBATCH * NTOK * CDIM;        // 3,145,728
  const int nq = 3 * CDIM * CDIM;             // 1,769,472
  const int np = CDIM * CDIM;                 //   589,824
  int i = blockIdx.x * 256 + threadIdx.x;
  if (i < nx)                 Xb[i]            = f2bf(x[i]);
  else if (i < nx + nq)       Wqkvb[i - nx]    = f2bf(qkvw[i - nx]);
  else if (i < nx + nq + np)  Wpb[i - nx - nq] = f2bf(projw[i - nx - nq]);
}

// ------------------------------------------------------------- qkv  GEMM --
// [4096,2304] = Xb[4096,768] @ Wqkv[2304,768]^T + b; epilogue scatters bf16
// into Q[bh][n][d], K[bh][n][d] and transposed Vt[bh][d][n].
__global__ __launch_bounds__(256) void k_qkv(const unsigned short* __restrict__ Xb,
                                             const unsigned short* __restrict__ Wqkvb,
                                             const float* __restrict__ qkvb,
                                             unsigned short* __restrict__ Qb,
                                             unsigned short* __restrict__ Kb,
                                             unsigned short* __restrict__ Vt) {
  const int wid  = blockIdx.x * 8 + (threadIdx.x >> 5);
  const int lane = threadIdx.x & 31;
  if (wid >= 256 * 36) return;
  const int tm = wid / 36, tc = wid % 36;
  const int r0 = tm * 16, c0 = tc * 64;
  const int nl = lane & 15, half = lane >> 4;

  const unsigned short* arow = Xb + (size_t)(r0 + nl) * CDIM;
  const unsigned short* brow[4];
#pragma unroll
  for (int s = 0; s < 4; ++s)
    brow[s] = Wqkvb + (size_t)(c0 + s * 16 + nl) * CDIM;

  f32x8 acc[4] = {{}, {}, {}, {}};
  gemm_tile_pipelined(arow, brow, CDIM, lane, acc);

  const int c64 = c0 / 64;                 // 64-col tile == one head slice
  const int sidx = c64 / NHEAD;            // 0:q 1:k 2:v (wave-uniform)
  const int h    = c64 % NHEAD;
#pragma unroll
  for (int s = 0; s < 4; ++s) {
    const int col  = c0 + s * 16 + nl;
    const int d    = s * 16 + nl;
    const float bias = qkvb[col];
#pragma unroll
    for (int r = 0; r < 8; ++r) {
      const int row = r0 + half * 8 + r;
      const int b = row >> 10, n = row & (NTOK - 1);
      const unsigned short bv = f2bf(acc[s][r] + bias);
      const size_t bh = (size_t)b * NHEAD + h;
      if (sidx == 0)      Qb[(bh * NTOK + n) * HDIM + d] = bv;
      else if (sidx == 1) Kb[(bh * NTOK + n) * HDIM + d] = bv;
      else                Vt[(bh * HDIM + d) * NTOK + n] = bv;
    }
  }
}

// ----------------------------------------------------------- scores GEMM --
// qk[bh][n][m] = Q[bh][n][:] . K[bh][m][:]  (K-dim = 64, f32 out)
__global__ __launch_bounds__(256) void k_scores(const unsigned short* __restrict__ Qb,
                                                const unsigned short* __restrict__ Kb,
                                                float* __restrict__ qk) {
  const int wid  = blockIdx.x * 8 + (threadIdx.x >> 5);
  const int lane = threadIdx.x & 31;
  if (wid >= NBH * 64 * 16) return;
  const int bh = wid / (64 * 16);
  const int rr = wid % (64 * 16);
  const int n0 = (rr / 16) * 16, m0 = (rr % 16) * 64;
  const int nl = lane & 15, half = lane >> 4;

  const unsigned short* arow = Qb + ((size_t)bh * NTOK + n0 + nl) * HDIM;
  const unsigned short* brow[4];
#pragma unroll
  for (int s = 0; s < 4; ++s)
    brow[s] = Kb + ((size_t)bh * NTOK + m0 + s * 16 + nl) * HDIM;

  f32x8 acc[4] = {{}, {}, {}, {}};
  gemm_tile_pipelined(arow, brow, HDIM, lane, acc);

#pragma unroll
  for (int s = 0; s < 4; ++s) {
#pragma unroll
    for (int r = 0; r < 8; ++r) {
      const int n = n0 + half * 8 + r;
      const int m = m0 + s * 16 + nl;
      qk[((size_t)bh * NTOK + n) * NTOK + m] = acc[s][r];
    }
  }
}

// ------------------------------------------- fused mix/sigmoid/softmax ----
// One block per (b,n): stages qk[b,:,n,:] (12x1024 f32 = 48 KB) in LDS, then
// for each output head o: head-mix + sigmoid (uncertainty), two-pass softmax
// (attn_mean), and attn = attn_mean + unc*r written as bf16 for the AV GEMM.
__global__ __launch_bounds__(256) void k_mix(const float* __restrict__ qk,
                                             const float* __restrict__ rten,
                                             const float* __restrict__ du_w,
                                             const float* __restrict__ du_b,
                                             float* __restrict__ attn_mean,
                                             float* __restrict__ unc,
                                             unsigned short* __restrict__ attn_bf) {
  __shared__ float sqk[NHEAD * NTOK];
  __shared__ float sred[256];
  __shared__ float sduw[NHEAD * NHEAD];
  __shared__ float sdub[NHEAD];
  const int tid = threadIdx.x;
  const int b = blockIdx.x >> 10;
  const int n = blockIdx.x & (NTOK - 1);
  if (tid < NHEAD * NHEAD) sduw[tid] = du_w[tid];
  if (tid < NHEAD)         sdub[tid] = du_b[tid];
  for (int idx = tid; idx < NHEAD * NTOK; idx += 256) {
    const int h = idx >> 10, m = idx & (NTOK - 1);
    sqk[idx] = qk[(((size_t)(b * NHEAD + h)) * NTOK + n) * NTOK + m];
  }
  __syncthreads();
  const float scale = 0.125f;               // hd^-0.5, hd=64
  for (int o = 0; o < NHEAD; ++o) {
    const float* row = sqk + o * NTOK;
    float lm = -3.0e38f;
    for (int m = tid; m < NTOK; m += 256) lm = fmaxf(lm, row[m]);
    sred[tid] = lm; __syncthreads();
    for (int s = 128; s > 0; s >>= 1) {
      if (tid < s) sred[tid] = fmaxf(sred[tid], sred[tid + s]);
      __syncthreads();
    }
    const float rmax = sred[0] * scale; __syncthreads();
    float ls = 0.f;
    for (int m = tid; m < NTOK; m += 256) ls += __expf(row[m] * scale - rmax);
    sred[tid] = ls; __syncthreads();
    for (int s = 128; s > 0; s >>= 1) {
      if (tid < s) sred[tid] += sred[tid + s];
      __syncthreads();
    }
    const float rinv = 1.0f / sred[0]; __syncthreads();
    const size_t base = (((size_t)(b * NHEAD + o)) * NTOK + n) * NTOK;
    for (int m = tid; m < NTOK; m += 256) {
      const float am = __expf(row[m] * scale - rmax) * rinv;
      float mix = sdub[o];
#pragma unroll
      for (int h = 0; h < NHEAD; ++h) mix += sduw[o * NHEAD + h] * sqk[h * NTOK + m];
      const float u = 1.0f / (1.0f + __expf(-mix));
      attn_mean[base + m] = am;
      unc[base + m]       = u;
      attn_bf[base + m]   = f2bf(am + u * rten[base + m]);
    }
  }
}

// --------------------------------------------------------------- AV GEMM --
// Hc[b,n, h*64+d] = sum_m attn[bh][n][m] * Vt[bh][d][m]   (K-dim = 1024)
__global__ __launch_bounds__(256) void k_av(const unsigned short* __restrict__ attn_bf,
                                            const unsigned short* __restrict__ Vt,
                                            unsigned short* __restrict__ Hc) {
  const int wid  = blockIdx.x * 8 + (threadIdx.x >> 5);
  const int lane = threadIdx.x & 31;
  if (wid >= NBH * 64) return;
  const int bh = wid / 64;
  const int n0 = (wid % 64) * 16;
  const int nl = lane & 15, half = lane >> 4;

  const unsigned short* arow = attn_bf + ((size_t)bh * NTOK + n0 + nl) * NTOK;
  const unsigned short* brow[4];
#pragma unroll
  for (int s = 0; s < 4; ++s)
    brow[s] = Vt + ((size_t)bh * HDIM + s * 16 + nl) * NTOK;

  f32x8 acc[4] = {{}, {}, {}, {}};
  gemm_tile_pipelined(arow, brow, NTOK, lane, acc);

  const int b = bh / NHEAD, h = bh % NHEAD;
#pragma unroll
  for (int s = 0; s < 4; ++s) {
    const int d = s * 16 + nl;
#pragma unroll
    for (int r = 0; r < 8; ++r) {
      const int n = n0 + half * 8 + r;
      Hc[((size_t)(b * NTOK + n)) * CDIM + h * HDIM + d] = f2bf(acc[s][r]);
    }
  }
}

// ------------------------------------------------------------- proj GEMM --
// out[4096,768] = Hc[4096,768] @ Wp[768,768]^T + b  (f32 out to d_out)
__global__ __launch_bounds__(256) void k_proj(const unsigned short* __restrict__ Hc,
                                              const unsigned short* __restrict__ Wpb,
                                              const float* __restrict__ projb,
                                              float* __restrict__ out) {
  const int wid  = blockIdx.x * 8 + (threadIdx.x >> 5);
  const int lane = threadIdx.x & 31;
  if (wid >= 256 * 12) return;
  const int r0 = (wid / 12) * 16, c0 = (wid % 12) * 64;
  const int nl = lane & 15, half = lane >> 4;

  const unsigned short* arow = Hc + (size_t)(r0 + nl) * CDIM;
  const unsigned short* brow[4];
#pragma unroll
  for (int s = 0; s < 4; ++s)
    brow[s] = Wpb + (size_t)(c0 + s * 16 + nl) * CDIM;

  f32x8 acc[4] = {{}, {}, {}, {}};
  gemm_tile_pipelined(arow, brow, CDIM, lane, acc);

#pragma unroll
  for (int s = 0; s < 4; ++s) {
    const int col = c0 + s * 16 + nl;
    const float bias = projb[col];
#pragma unroll
    for (int r = 0; r < 8; ++r) {
      const int row = r0 + half * 8 + r;
      out[(size_t)row * CDIM + col] = acc[s][r] + bias;
    }
  }
}

// ------------------------------------------------------------------ host --
extern "C" void kernel_launch(void* const* d_in, const int* in_sizes, int n_in,
                              void* d_out, int out_size, void* d_ws, size_t ws_size,
                              hipStream_t stream) {
  const float* x      = (const float*)d_in[0];
  const float* rten   = (const float*)d_in[1];
  const float* qkv_w  = (const float*)d_in[2];
  const float* qkv_b  = (const float*)d_in[3];
  const float* proj_w = (const float*)d_in[4];
  const float* proj_b = (const float*)d_in[5];
  const float* du_w   = (const float*)d_in[6];
  const float* du_b   = (const float*)d_in[7];

  float* out       = (float*)d_out;                                   // [4,1024,768]
  float* attn_mean = out + (size_t)BBATCH * NTOK * CDIM;              // [4,12,1024,1024]
  float* unc       = attn_mean + (size_t)NBH * NTOK * NTOK;           // [4,12,1024,1024]

  // Workspace carve-up (all chunks 256B-aligned; ~338 MB total).
  char* ws = (char*)d_ws;
  unsigned short* Xb    = (unsigned short*)ws;  ws += (size_t)BBATCH * NTOK * CDIM * 2;
  unsigned short* Wqkvb = (unsigned short*)ws;  ws += (size_t)3 * CDIM * CDIM * 2;
  unsigned short* Wpb   = (unsigned short*)ws;  ws += (size_t)CDIM * CDIM * 2;
  unsigned short* Qb    = (unsigned short*)ws;  ws += (size_t)NBH * NTOK * HDIM * 2;
  unsigned short* Kb    = (unsigned short*)ws;  ws += (size_t)NBH * NTOK * HDIM * 2;
  unsigned short* Vt    = (unsigned short*)ws;  ws += (size_t)NBH * NTOK * HDIM * 2;
  float*          qk    = (float*)ws;           ws += (size_t)NBH * NTOK * NTOK * 4;
  unsigned short* Abf   = (unsigned short*)ws;  ws += (size_t)NBH * NTOK * NTOK * 2;
  unsigned short* Hc    = (unsigned short*)ws;

  (void)in_sizes; (void)n_in; (void)out_size; (void)ws_size;

  k_convert<<<21504, 256, 0, stream>>>(x, qkv_w, proj_w, Xb, Wqkvb, Wpb);
  k_qkv   <<<1152,  256, 0, stream>>>(Xb, Wqkvb, qkv_b, Qb, Kb, Vt);
  k_scores<<<6144,  256, 0, stream>>>(Qb, Kb, qk);
  k_mix   <<<BBATCH * NTOK, 256, 0, stream>>>(qk, rten, du_w, du_b,
                                              attn_mean, unc, Abf);
  k_av    <<<384,   256, 0, stream>>>(Abf, Vt, Hc);
  k_proj  <<<384,   256, 0, stream>>>(Hc, Wpb, proj_b, out);
}
